// ResidualVQ_33818572488896
// MI455X (gfx1250) — compile-verified, matrix operand order checked by reading, not run on hardware
//
#include <hip/hip_runtime.h>
#include <hip/hip_bf16.h>

// ---------------------------------------------------------------------------
// Residual VQ for MI455X (gfx1250):
//   distance GEMM via v_wmma_f32_16x16x32_bf16, codebook tiles staged in LDS
//   by the Tensor Data Mover (tensor_load_to_lds + s_wait_tensorcnt),
//   double-buffered so the DMA overlaps the WMMA stream.
// ---------------------------------------------------------------------------

typedef __attribute__((ext_vector_type(16))) __bf16      v16bf;
typedef __attribute__((ext_vector_type(8)))  float        v8f;
typedef __attribute__((ext_vector_type(4)))  unsigned int v4u;
typedef __attribute__((ext_vector_type(4)))  unsigned int u32x4;
typedef __attribute__((ext_vector_type(8)))  int          i32x8;
typedef __attribute__((ext_vector_type(4)))  int          i32x4;

union Frag16 {
    v16bf v;
    v4u   u4[2];
};

static constexpr int Bn   = 8;
static constexpr int Nn   = 4096;
static constexpr int Dn   = 256;
static constexpr int Qn   = 8;
static constexpr int Cn   = 1024;
static constexpr int ROWS = Bn * Nn;            // 32768
static constexpr int NWAVES = 8;                // 256 threads, wave32
static constexpr int TMB  = NWAVES * 16;        // 128 rows per block
static constexpr int CHUNK = 64;                // codes staged per TDM transfer
static constexpr int NCHUNK = Cn / CHUNK;       // 16
static constexpr int LDA  = Dn + 8;             // ushort stride: 528B (16B mult)
static constexpr int BUFB = CHUNK * LDA * 2;    // bytes per B buffer (33792)

#if defined(__has_builtin)
#if __has_builtin(__builtin_amdgcn_tensor_load_to_lds)
#define RVQ_HAS_TDM 1
#endif
#endif
#ifndef RVQ_HAS_TDM
#define RVQ_HAS_TDM 0
#endif

__device__ __forceinline__ unsigned short f32_to_bf16(float f) {
    union { float f; unsigned u; } x; x.f = f;
    unsigned r = x.u + 0x7fffu + ((x.u >> 16) & 1u);   // round-to-nearest-even
    return (unsigned short)(r >> 16);
}

// Precompute bf16 codebooks + half squared norms (0.5*||c||^2).
__global__ __launch_bounds__(256)
void rvq_prep(const float* __restrict__ cb,            // [Q*C, D] f32
              unsigned short* __restrict__ cbBf,       // [Q*C, D] bf16
              float* __restrict__ hn)                  // [Q*C]
{
    __shared__ float sPart[NWAVES];
    const int row = blockIdx.x;                        // 0..Q*C-1
    const int t   = threadIdx.x;                       // 0..255 == D
    float v = cb[(size_t)row * Dn + t];
    cbBf[(size_t)row * Dn + t] = f32_to_bf16(v);
    float ss = v * v;
    #pragma unroll
    for (int off = 16; off > 0; off >>= 1)
        ss += __shfl_down(ss, off, 32);
    if ((t & 31) == 0) sPart[t >> 5] = ss;
    __syncthreads();
    if (t == 0) {
        float tot = 0.f;
        #pragma unroll
        for (int i = 0; i < NWAVES; ++i) tot += sPart[i];
        hn[row] = 0.5f * tot;
    }
}

#if RVQ_HAS_TDM
// Issue one TDM transfer: 64 codebook rows (bf16, 512B each) -> LDS with a
// 16B pad after every row (stride 528B) so ds_load_b128 B-fragment reads are
// bank-conflict-free.  D# layout per CDNA5 ISA ch.8.
__device__ __forceinline__ void tdm_load_chunk(const unsigned short* gsrc,
                                               unsigned lds_off) {
    unsigned long long ga = (unsigned long long)gsrc;
    u32x4 g0;
    g0[0] = 1u;                                        // count=1, user mode
    g0[1] = lds_off;                                   // lds_addr
    g0[2] = (unsigned)(ga & 0xffffffffu);              // global_addr[31:0]
    g0[3] = (unsigned)((ga >> 32) & 0x1ffffffu)        // global_addr[56:32]
          | (2u << 30);                                // type = 2 (image)
    i32x8 g1;
    g1[0] = (int)((1u << 16)                           // data_size = 2 bytes
          |        (1u << 20)                          // pad_enable
          |        (6u << 22)                          // pad_interval: 128 dw
          |        (3u << 25));                        // pad_amount: 4 dwords
    g1[1] = (int)(256u << 16);                         // tensor_dim0 = 256
    g1[2] = (int)(64u  << 16);                         // tensor_dim1 = 64
    g1[3] = (int)(256u << 16);                         // tile_dim0 = 256
    g1[4] = (int)64;                                   // tile_dim1 = 64
    g1[5] = (int)256;                                  // tensor_dim0_stride
    g1[6] = 0;
    g1[7] = 0;
    i32x4 gz = {0, 0, 0, 0};
#if __clang_major__ >= 23
    i32x8 gz8 = {0, 0, 0, 0, 0, 0, 0, 0};
    __builtin_amdgcn_tensor_load_to_lds(g0, g1, gz, gz, gz8, 0);
#else
    __builtin_amdgcn_tensor_load_to_lds(g0, g1, gz, gz, 0);
#endif
}
#endif

// One RVQ step.  Block = 8 waves; wave w owns rows [wave*16, wave*16+16) of a
// 128-row tile and scans all C=1024 codes.  Codebook chunks are staged in LDS
// (TDM double-buffered) and shared by all waves.
__global__ __launch_bounds__(256)
void rvq_step(const float* __restrict__ resIn,         // [ROWS, D] (x on step 0)
              float* __restrict__ resOut,              // [ROWS, D] (in-place ok)
              const unsigned short* __restrict__ cbBf, // [C, D] bf16, this q
              const float* __restrict__ hn,            // [C], this q
              const float* __restrict__ cbF32,         // [C, D] f32, this q
              float* __restrict__ quantOut,            // [ROWS, D]
              int* __restrict__ idxOut,                // [ROWS, Q]
              int q, int firstStep)
{
    __shared__ __align__(16) unsigned short sA[TMB * LDA];        // 66 KB
    __shared__ __align__(16) unsigned short sB[2][CHUNK * LDA];   // 66 KB
    __shared__ int sFinal[TMB];

    const int t    = threadIdx.x;
    const int lane = t & 31;
    const int wave = t >> 5;
    const long rowBase = (long)blockIdx.x * TMB;

    // ---- Phase 1: stage 128x256 residual tile as bf16 into LDS ------------
    for (int i = t; i < TMB * (Dn / 4); i += 256) {
        const int row = i >> 6;                 // / (Dn/4)
        const int c4  = (i & 63) << 2;
        float4 v = *(const float4*)(resIn + (rowBase + row) * Dn + c4);
        sA[row * LDA + c4 + 0] = f32_to_bf16(v.x);
        sA[row * LDA + c4 + 1] = f32_to_bf16(v.y);
        sA[row * LDA + c4 + 2] = f32_to_bf16(v.z);
        sA[row * LDA + c4 + 3] = f32_to_bf16(v.w);
    }
    __syncthreads();

    // ---- A fragments (ISA 7.12.2): lane l: M=l&15, kb=(l>>4)*8 ------------
    const int m16 = lane & 15;
    const int kb  = (lane >> 4) * 8;
    Frag16 a[8];
    #pragma unroll
    for (int kc = 0; kc < 8; ++kc) {
        const int ab = (wave * 16 + m16) * LDA + kc * 32 + kb;
        a[kc].u4[0] = *(const v4u*)&sA[ab];
        a[kc].u4[1] = *(const v4u*)&sA[ab + 16];
    }

    float rs[8];
    int   ri[8];
    #pragma unroll
    for (int r = 0; r < 8; ++r) { rs[r] = -3.4e38f; ri[r] = 0; }

#if RVQ_HAS_TDM
    const unsigned sB_off = (unsigned)(unsigned long long)(const void*)&sB[0][0];
    if (wave == 0)
        tdm_load_chunk(cbBf, sB_off);
#endif

    for (int ch = 0; ch < NCHUNK; ++ch) {
        const int buf = ch & 1;
#if RVQ_HAS_TDM
        if (wave == 0)
            __builtin_amdgcn_s_wait_tensorcnt(0);     // chunk ch landed
        __syncthreads();                               // visible to all waves
        if (wave == 0 && ch + 1 < NCHUNK)
            tdm_load_chunk(cbBf + (size_t)(ch + 1) * CHUNK * Dn,
                           sB_off + (unsigned)((ch + 1) & 1) * BUFB);
#else
        __syncthreads();                               // buf free for reuse
        // Cooperative fallback copy: 64 rows of 512B, padded stride in LDS.
        for (int i = t; i < CHUNK * (Dn / 8); i += 256) {
            const int row = i >> 5;                    // / (Dn/8)
            const int c8  = (i & 31) << 3;
            __builtin_prefetch(cbBf + (size_t)ch * CHUNK * Dn + row * Dn + c8, 0, 0);
            v4u v = *(const v4u*)(cbBf + (size_t)ch * CHUNK * Dn + row * Dn + c8);
            *(v4u*)&sB[buf][row * LDA + c8] = v;
        }
        __syncthreads();
#endif
        // ---- 4 C-tiles x 8 K-chunks of WMMA from this LDS chunk ----------
        #pragma unroll
        for (int ctl = 0; ctl < CHUNK / 16; ++ctl) {
            const int code = ch * CHUNK + ctl * 16 + m16;  // this lane's col
            const unsigned short* brow = &sB[buf][(ctl * 16 + m16) * LDA];
            v8f acc = {};
            #pragma unroll
            for (int kc = 0; kc < 8; ++kc) {
                Frag16 b;
                b.u4[0] = *(const v4u*)(brow + kc * 32 + kb);
                b.u4[1] = *(const v4u*)(brow + kc * 32 + 16 + kb);
                acc = __builtin_amdgcn_wmma_f32_16x16x32_bf16(
                    false, a[kc].v, false, b.v, (short)0, acc, false, false);
            }
            const float h = hn[code];
            // C/D layout: lane l col n=l&15; vgpr r -> row (l<16 ? r : 8+r)
            #pragma unroll
            for (int r = 0; r < 8; ++r) {
                float s = acc[r] - h;
                if (s > rs[r] || (s == rs[r] && code < ri[r])) {
                    rs[r] = s; ri[r] = code;
                }
            }
        }
        __syncthreads();   // all waves done with buf before TDM reuses it
    }

    // ---- argmax across the 16 lanes of each half-wave ---------------------
    #pragma unroll
    for (int mask = 1; mask < 16; mask <<= 1) {
        #pragma unroll
        for (int r = 0; r < 8; ++r) {
            float os = __shfl_xor(rs[r], mask, 32);
            int   oi = __shfl_xor(ri[r], mask, 32);
            if (os > rs[r] || (os == rs[r] && oi < ri[r])) { rs[r] = os; ri[r] = oi; }
        }
    }
    if ((lane & 15) == 0) {
        const int mb = (lane >> 4) * 8;
        #pragma unroll
        for (int r = 0; r < 8; ++r) {
            const int m = wave * 16 + mb + r;
            sFinal[m] = ri[r];
            idxOut[(rowBase + m) * Qn + q] = ri[r];
        }
    }
    __syncthreads();

    // ---- Phase 3: gather f32 code, accumulate output, update residual -----
    for (int i = t; i < TMB * (Dn / 4); i += 256) {
        const int row = i >> 6;
        const int c4  = (i & 63) << 2;
        const int bi  = sFinal[row];
        const long g  = (rowBase + row) * Dn + c4;
        float4 cv = *(const float4*)(cbF32 + (size_t)bi * Dn + c4);
        float4 rv = *(const float4*)(resIn + g);
        float4 qv;
        if (firstStep) {
            qv = cv;
        } else {
            qv = *(const float4*)(quantOut + g);
            qv.x += cv.x; qv.y += cv.y; qv.z += cv.z; qv.w += cv.w;
        }
        *(float4*)(quantOut + g) = qv;
        float4 nv;
        nv.x = rv.x - cv.x; nv.y = rv.y - cv.y;
        nv.z = rv.z - cv.z; nv.w = rv.w - cv.w;
        *(float4*)(resOut + g) = nv;
    }
}

extern "C" void kernel_launch(void* const* d_in, const int* in_sizes, int n_in,
                              void* d_out, int out_size, void* d_ws, size_t ws_size,
                              hipStream_t stream) {
    (void)in_sizes; (void)n_in; (void)out_size; (void)ws_size;

    const float* x  = (const float*)d_in[0];   // [B, N, D]
    const float* cb = (const float*)d_in[1];   // [Q, C, D]

    float* quantOut = (float*)d_out;                                   // [B,N,D]
    int*   idxOut   = (int*)((float*)d_out + (size_t)ROWS * Dn);       // [B,N,Q]

    // Workspace: bf16 codebooks (4MB) | half-norms (32KB) | residual (32MB)
    unsigned short* cbBf = (unsigned short*)d_ws;
    float* hn  = (float*)(cbBf + (size_t)Qn * Cn * Dn);
    float* res = hn + (size_t)Qn * Cn;

    rvq_prep<<<Qn * Cn, 256, 0, stream>>>(cb, cbBf, hn);

    const float* rin = x;
    for (int q = 0; q < Qn; ++q) {
        rvq_step<<<ROWS / TMB, 256, 0, stream>>>(
            rin, res,
            cbBf + (size_t)q * Cn * Dn,
            hn   + (size_t)q * Cn,
            cb   + (size_t)q * Cn * Dn,
            quantOut, idxOut, q, q == 0 ? 1 : 0);
        rin = res;   // residual updated in place from step 1 onward
    }
}
